// SelfAttention_2362232012882
// MI455X (gfx1250) — matrix operand order repaired and checked by reading, not compiled, not run on hardware
//
#include <hip/hip_runtime.h>
#include <stdint.h>

#define BATCH 4
#define SEQ   2048
#define DIM   1024
#define NH    16
#define HDIM  64
#define MROWS (BATCH*SEQ)   // 8192

typedef __attribute__((ext_vector_type(16))) __bf16 bf16x16;
typedef __attribute__((ext_vector_type(8)))  __bf16 bf16x8;
typedef __attribute__((ext_vector_type(8)))  float  f32x8;

__device__ __forceinline__ uint16_t f32_to_bf16_bits(float f) {
  uint32_t u = __builtin_bit_cast(uint32_t, f);
  u += 0x7FFFu + ((u >> 16) & 1u);           // round-to-nearest-even
  return (uint16_t)(u >> 16);
}

__device__ __forceinline__ f32x8 zero8() {
  f32x8 z = {0.f,0.f,0.f,0.f,0.f,0.f,0.f,0.f};
  return z;
}

// Build a 16-element bf16 fragment from two contiguous 8-element (16B) runs.
__device__ __forceinline__ bf16x16 frag16(const uint16_t* p0, const uint16_t* p1) {
  bf16x8 lo = *reinterpret_cast<const bf16x8*>(p0);
  bf16x8 hi = *reinterpret_cast<const bf16x8*>(p1);
  return __builtin_shufflevector(lo, hi, 0,1,2,3,4,5,6,7,8,9,10,11,12,13,14,15);
}

__device__ __forceinline__ f32x8 wmma_bf16(bf16x16 a, bf16x16 b, f32x8 c) {
  // (neg_a, A, neg_b, B, c_mod, C, reuse_a, reuse_b)
  return __builtin_amdgcn_wmma_f32_16x16x32_bf16(false, a, false, b, (short)0, c,
                                                 false, false);
}

// ds_swizzle_b32 xor-swap within 32-lane groups (masks < 16 stay inside each
// 16-lane row group). offset = xor<<10 | or<<5 | and(0x1f).
template <int IMM>
__device__ __forceinline__ float dswz(float v) {
  int r = __builtin_amdgcn_ds_swizzle(__builtin_bit_cast(int, v), IMM);
  return __builtin_bit_cast(float, r);
}
__device__ __forceinline__ float rowmax16(float v) {
  v = fmaxf(v, dswz<0x041F>(v));   // SWAPX1
  v = fmaxf(v, dswz<0x081F>(v));   // SWAPX2
  v = fmaxf(v, dswz<0x101F>(v));   // SWAPX4
  v = fmaxf(v, dswz<0x201F>(v));   // SWAPX8
  return v;
}
__device__ __forceinline__ float rowsum16(float v) {
  v += dswz<0x041F>(v);
  v += dswz<0x081F>(v);
  v += dswz<0x101F>(v);
  v += dswz<0x201F>(v);
  return v;
}

// ---------------------------------------------------------------- convert ---
__global__ __launch_bounds__(256) void cvt_f32_bf16(const float* __restrict__ src,
                                                    uint16_t* __restrict__ dst,
                                                    int n4) {
  int i = blockIdx.x * blockDim.x + threadIdx.x;
  if (i >= n4) return;
  const float4 f = reinterpret_cast<const float4*>(src)[i];
  ushort4 r;
  r.x = f32_to_bf16_bits(f.x); r.y = f32_to_bf16_bits(f.y);
  r.z = f32_to_bf16_bits(f.z); r.w = f32_to_bf16_bits(f.w);
  reinterpret_cast<ushort4*>(dst)[i] = r;
}

// ------------------------------------------------------------------- GEMM ---
// C[M,N] = A[M,K] * W[N,K]^T + bias. MODE 0: bf16 out in [B,H,S,HD] layout.
// MODE 1: fp32 out, plain [M,N] row-major.
template <int MODE>
__global__ __launch_bounds__(256) void gemm_bf16_wmma(
    const uint16_t* __restrict__ A,
    const uint16_t* __restrict__ W,
    const float*    __restrict__ bias,
    void* __restrict__ outp,
    int M, int N, int K) {
  (void)M;
  __shared__ __attribute__((aligned(16))) uint16_t As[64][40];    // pad: 80B rows
  __shared__ __attribute__((aligned(16))) uint16_t Bs[128][40];

  const int tid  = threadIdx.x;
  const int lane = tid & 31;
  const int wave = tid >> 5;
  const int hw   = lane >> 4;          // half-wave select
  const int ln   = lane & 15;
  const int wm   = wave >> 2;          // 0..1  (wave M tile)
  const int wn   = wave & 3;           // 0..3  (wave N tile)

  const int m0 = blockIdx.y * 64;
  const int n0 = blockIdx.x * 128;

  f32x8 acc[2][2];
#pragma unroll
  for (int i = 0; i < 2; ++i)
#pragma unroll
    for (int j = 0; j < 2; ++j) acc[i][j] = zero8();

  const int arow = tid >> 2, acol = (tid & 3) * 8;    // 64 x 32 A tile
  const int brow = tid >> 1, bcol = (tid & 1) * 16;   // 128 x 32 B tile
  const int koff = 8 * hw;

  for (int k0 = 0; k0 < K; k0 += 32) {
    *reinterpret_cast<uint4*>(&As[arow][acol]) =
        *reinterpret_cast<const uint4*>(&A[(size_t)(m0 + arow) * K + k0 + acol]);
    *reinterpret_cast<uint4*>(&Bs[brow][bcol]) =
        *reinterpret_cast<const uint4*>(&W[(size_t)(n0 + brow) * K + k0 + bcol]);
    *reinterpret_cast<uint4*>(&Bs[brow][bcol + 8]) =
        *reinterpret_cast<const uint4*>(&W[(size_t)(n0 + brow) * K + k0 + bcol + 8]);
    __syncthreads();

    bf16x16 af[2], bfm[2];
#pragma unroll
    for (int i = 0; i < 2; ++i) {
      const uint16_t* r = As[wm * 32 + i * 16 + ln];
      af[i] = frag16(r + 16 * hw, r + 16 * hw + 8);        // A layout: 16 contig K
    }
#pragma unroll
    for (int j = 0; j < 2; ++j) {
      const uint16_t* r = Bs[wn * 32 + j * 16 + ln];
      bfm[j] = frag16(r + 16 * hw, r + 16 * hw + 8);       // B layout: 16 contig K
    }
#pragma unroll
    for (int i = 0; i < 2; ++i)
#pragma unroll
      for (int j = 0; j < 2; ++j)
        acc[i][j] = wmma_bf16(af[i], bfm[j], acc[i][j]);
    __syncthreads();
  }

#pragma unroll
  for (int i = 0; i < 2; ++i) {
#pragma unroll
    for (int j = 0; j < 2; ++j) {
      const int n = n0 + wn * 32 + j * 16 + ln;
      const float bcolv = bias[n];
#pragma unroll
      for (int e = 0; e < 8; ++e) {
        const int row = m0 + wm * 32 + i * 16 + e + 8 * hw;
        const float val = acc[i][j][e] + bcolv;
        if (MODE == 0) {
          const int bidx = row >> 11, s = row & (SEQ - 1);
          const int head = n >> 6, hd = n & (HDIM - 1);
          reinterpret_cast<uint16_t*>(outp)[
              (((size_t)bidx * NH + head) * SEQ + s) * HDIM + hd] =
              f32_to_bf16_bits(val);
        } else {
          reinterpret_cast<float*>(outp)[(size_t)row * N + n] = val;
        }
      }
    }
  }
}

// -------------------------------------------------------- flash attention ---
__global__ __launch_bounds__(256) void attn_fwd_wmma(
    const uint16_t* __restrict__ Q,    // [B*H, S, HD] bf16
    const uint16_t* __restrict__ Km,
    const uint16_t* __restrict__ Vm,
    uint16_t* __restrict__ ctx) {      // [B*S, D] bf16
  __shared__ __attribute__((aligned(16))) uint16_t Ks[32][72];       // keys x dims
  __shared__ __attribute__((aligned(16))) uint16_t Vs[64][40];       // dims x keys
  __shared__ __attribute__((aligned(16))) uint16_t Ps[8][16][40];    // per-wave P

  const int tid  = threadIdx.x;
  const int lane = tid & 31, wave = tid >> 5;
  const int hw   = lane >> 4, ln = lane & 15;
  const int bh   = blockIdx.y;
  const int q0   = blockIdx.x * 128 + wave * 16;

  const uint16_t* qbase = Q  + (size_t)bh * SEQ * HDIM;
  const uint16_t* kbase = Km + (size_t)bh * SEQ * HDIM;
  const uint16_t* vbase = Vm + (size_t)bh * SEQ * HDIM;

  // Q fragments live in registers for the whole pass (16 rows x 64 dims).
  bf16x16 qa[2];
  {
    const uint16_t* qr = qbase + (size_t)(q0 + ln) * HDIM;
    const int koff = 8 * hw;
    qa[0] = frag16(qr + 2 * koff,      qr + 2 * koff + 8);
    qa[1] = frag16(qr + 32 + 2 * koff, qr + 32 + 2 * koff + 8);
  }

  f32x8 o[4];
#pragma unroll
  for (int j = 0; j < 4; ++j) o[j] = zero8();
  float mrow[8], lrow[8];
#pragma unroll
  for (int e = 0; e < 8; ++e) { mrow[e] = -1e30f; lrow[e] = 0.f; }

  // softmax kept in exp2 domain: fold (1/sqrt(HD)) * log2(e) into one constant
  const float cscale = 0.125f * 1.44269504088896340736f;
  const int krow = tid >> 3, kcol = (tid & 7) * 8;
  const int vkey = tid >> 3, vdim = (tid & 7) * 8;

  for (int kt = 0; kt < SEQ / 32; ++kt) {
    const int kb0 = kt * 32;
    // K tile: 32 keys x 64 dims, row-major in LDS
    *reinterpret_cast<uint4*>(&Ks[krow][kcol]) =
        *reinterpret_cast<const uint4*>(&kbase[(size_t)(kb0 + krow) * HDIM + kcol]);
    // V tile transposed: Vs[dim][key] so PV B-fragments are contiguous
    {
      uint4 raw = *reinterpret_cast<const uint4*>(
          &vbase[(size_t)(kb0 + vkey) * HDIM + vdim]);
      const uint16_t* el = reinterpret_cast<const uint16_t*>(&raw);
#pragma unroll
      for (int t = 0; t < 8; ++t) Vs[vdim + t][vkey] = el[t];
    }
    if (kt + 1 < SEQ / 32) {   // global_prefetch_b8 for next tile
      __builtin_prefetch(&kbase[(size_t)(kb0 + 32 + krow) * HDIM + kcol], 0, 1);
      __builtin_prefetch(&vbase[(size_t)(kb0 + 32 + vkey) * HDIM + vdim], 0, 1);
    }
    __syncthreads();

    // scores: 16 q rows x 32 keys (2 x 16-key subtiles, 2 WMMAs each)
    f32x8 sc[2];
#pragma unroll
    for (int st = 0; st < 2; ++st) {
      const uint16_t* krw = Ks[st * 16 + ln];
      bf16x16 b0 = frag16(krw + 16 * hw,      krw + 16 * hw + 8);
      bf16x16 b1 = frag16(krw + 32 + 16 * hw, krw + 32 + 16 * hw + 8);
      f32x8 s = zero8();
      s = wmma_bf16(qa[0], b0, s);
      s = wmma_bf16(qa[1], b1, s);
      sc[st] = s;
    }

    // online softmax: 8 independent row reductions, fully unrolled so the
    // ds_swizzle lane-permutes pipeline across rows
    float alpha[8];
#pragma unroll
    for (int e = 0; e < 8; ++e) {
      const float s0 = sc[0][e] * cscale, s1 = sc[1][e] * cscale;
      const float t  = rowmax16(fmaxf(s0, s1));
      const float mn = fmaxf(mrow[e], t);
      alpha[e] = __builtin_amdgcn_exp2f(mrow[e] - mn);
      mrow[e] = mn;
      const float p0 = __builtin_amdgcn_exp2f(s0 - mn);
      const float p1 = __builtin_amdgcn_exp2f(s1 - mn);
      lrow[e] = lrow[e] * alpha[e] + rowsum16(p0 + p1);
      const int r = e + 8 * hw;
      Ps[wave][r][ln]      = f32_to_bf16_bits(p0);   // D-layout -> A-layout via LDS
      Ps[wave][r][16 + ln] = f32_to_bf16_bits(p1);
    }
#pragma unroll
    for (int j = 0; j < 4; ++j)
#pragma unroll
      for (int e = 0; e < 8; ++e) o[j][e] *= alpha[e];

    // per-wave LDS round-trip: wait on the DS counter before reading P back
    asm volatile("s_wait_dscnt 0" ::: "memory");

    // PV: O(16x64) += P(16x32) x V(32x64), 4 WMMAs
    {
      const uint16_t* pr = Ps[wave][ln];
      const int koff = 8 * hw;
      bf16x16 ap = frag16(pr + 2 * koff, pr + 2 * koff + 8);
#pragma unroll
      for (int j = 0; j < 4; ++j) {
        const uint16_t* vr = Vs[j * 16 + ln];
        bf16x16 vbf = frag16(vr + 16 * hw, vr + 16 * hw + 8);
        o[j] = wmma_bf16(ap, vbf, o[j]);
      }
    }
    __syncthreads();
  }

  // normalize and emit ctx in [B*S, D] bf16 (ready for output projection)
  const int b = bh >> 4, head = bh & 15;
  float inv[8];
#pragma unroll
  for (int e = 0; e < 8; ++e) inv[e] = __builtin_amdgcn_rcpf(lrow[e]);
#pragma unroll
  for (int j = 0; j < 4; ++j) {
    const int col = head * HDIM + j * 16 + ln;
#pragma unroll
    for (int e = 0; e < 8; ++e) {
      const int srow = q0 + e + 8 * hw;
      ctx[((size_t)b * SEQ + srow) * DIM + col] = f32_to_bf16_bits(o[j][e] * inv[e]);
    }
  }
}

// ----------------------------------------------------------------- launch ---
extern "C" void kernel_launch(void* const* d_in, const int* in_sizes, int n_in,
                              void* d_out, int out_size, void* d_ws, size_t ws_size,
                              hipStream_t stream) {
  (void)in_sizes; (void)n_in; (void)out_size; (void)ws_size;
  const float* x  = (const float*)d_in[0];
  const float* Wq = (const float*)d_in[1];
  const float* bq = (const float*)d_in[2];
  const float* Wk = (const float*)d_in[3];
  const float* bk = (const float*)d_in[4];
  const float* Wv = (const float*)d_in[5];
  const float* bv = (const float*)d_in[6];
  const float* Wo = (const float*)d_in[7];
  const float* bo = (const float*)d_in[8];

  char* ws = (char*)d_ws;
  const size_t MB = 1024ull * 1024ull;
  uint16_t* xb  = (uint16_t*)(ws);             // 16 MiB: x bf16
  uint16_t* wqb = (uint16_t*)(ws + 16 * MB);   //  2 MiB each: weights bf16
  uint16_t* wkb = (uint16_t*)(ws + 18 * MB);
  uint16_t* wvb = (uint16_t*)(ws + 20 * MB);
  uint16_t* wob = (uint16_t*)(ws + 22 * MB);
  uint16_t* qb  = (uint16_t*)(ws + 24 * MB);   // 16 MiB each: q/k/v [B,H,S,HD]
  uint16_t* kb  = (uint16_t*)(ws + 40 * MB);
  uint16_t* vb  = (uint16_t*)(ws + 56 * MB);
  uint16_t* cb  = (uint16_t*)(ws + 72 * MB);   // 16 MiB: ctx -> total 88 MiB

  const int nx = BATCH * SEQ * DIM;   // 8,388,608
  const int nw = DIM * DIM;           // 1,048,576
  cvt_f32_bf16<<<nx / 4 / 256, 256, 0, stream>>>(x,  xb,  nx / 4);
  cvt_f32_bf16<<<nw / 4 / 256, 256, 0, stream>>>(Wq, wqb, nw / 4);
  cvt_f32_bf16<<<nw / 4 / 256, 256, 0, stream>>>(Wk, wkb, nw / 4);
  cvt_f32_bf16<<<nw / 4 / 256, 256, 0, stream>>>(Wv, wvb, nw / 4);
  cvt_f32_bf16<<<nw / 4 / 256, 256, 0, stream>>>(Wo, wob, nw / 4);

  dim3 gg(DIM / 128, MROWS / 64);
  gemm_bf16_wmma<0><<<gg, 256, 0, stream>>>(xb, wqb, bq, qb, MROWS, DIM, DIM);
  gemm_bf16_wmma<0><<<gg, 256, 0, stream>>>(xb, wkb, bk, kb, MROWS, DIM, DIM);
  gemm_bf16_wmma<0><<<gg, 256, 0, stream>>>(xb, wvb, bv, vb, MROWS, DIM, DIM);

  attn_fwd_wmma<<<dim3(SEQ / 128, BATCH * NH), 256, 0, stream>>>(qb, kb, vb, cb);

  gemm_bf16_wmma<1><<<gg, 256, 0, stream>>>(cb, wob, bo, d_out, MROWS, DIM, DIM);
}
